// MutiHeadAttention_90220083020249
// MI455X (gfx1250) — matrix operand order repaired and checked by reading, not compiled, not run on hardware
//
#include <hip/hip_runtime.h>
#include <math.h>

// ---------------------------------------------------------------------------
// Types for CDNA5 WMMA (wave32): v_wmma_f32_16x16x32_bf16
// ---------------------------------------------------------------------------
typedef __attribute__((ext_vector_type(16))) __bf16 v16bf;
typedef __attribute__((ext_vector_type(8)))  float  v8f;

union V16U {
    v16bf          v;
    unsigned short u[16];
    uint4          q[2];   // two 16-byte chunks
};

__device__ __forceinline__ unsigned short f32_to_bf16(float f) {
    unsigned int x = __float_as_uint(f);
    x += 0x7FFFu + ((x >> 16) & 1u);   // round-to-nearest-even
    return (unsigned short)(x >> 16);
}

// ---------------------------------------------------------------------------
// gfx1250 async global->LDS copies (ASYNCcnt-tracked, bypass VGPRs), with a
// guarded fallback to synchronous copies if the builtins are absent.
// Probe-confirmed signature: (v4i AS1* src, v4i AS3* dst, imm offset, imm cpol).
// ---------------------------------------------------------------------------
#if defined(__has_builtin)
#  if __has_builtin(__builtin_amdgcn_global_load_async_to_lds_b128)
#    define HAVE_ASYNC_LDS 1
#  endif
#  if __has_builtin(__builtin_amdgcn_s_wait_asynccnt)
#    define HAVE_WAIT_ASYNC 1
#  endif
#endif

typedef int v4i_vec __attribute__((vector_size(16)));

__device__ __forceinline__ void wait_asynccnt0() {
#if defined(HAVE_WAIT_ASYNC)
    __builtin_amdgcn_s_wait_asynccnt(0);
#elif defined(HAVE_ASYNC_LDS)
    asm volatile("s_wait_asynccnt 0x0" ::: "memory");
#endif
}

// copy 32 bytes (16 bf16) global -> LDS
__device__ __forceinline__ void async_copy_32B(const unsigned short* g, unsigned short* l) {
#if defined(HAVE_ASYNC_LDS)
    __builtin_amdgcn_global_load_async_to_lds_b128(
        (__attribute__((address_space(1))) v4i_vec*)g,
        (__attribute__((address_space(3))) v4i_vec*)l, 0, 0);
    __builtin_amdgcn_global_load_async_to_lds_b128(
        (__attribute__((address_space(1))) v4i_vec*)(g + 8),
        (__attribute__((address_space(3))) v4i_vec*)(l + 8), 0, 0);
#else
    ((uint4*)l)[0] = ((const uint4*)g)[0];
    ((uint4*)l)[1] = ((const uint4*)g)[1];
#endif
}

// Problem constants
#define BB 2
#define SS 2048
#define DD 1024
#define HH 16
#define ND 64
#define GM 4096   // B*S
#define GN 1024   // D
#define GK 1024   // D

// ---------------------------------------------------------------------------
// f32 -> bf16 conversion (grid-stride)
// ---------------------------------------------------------------------------
__global__ void cvt_bf16_kernel(const float* __restrict__ in,
                                unsigned short* __restrict__ out, int n) {
    int i = blockIdx.x * blockDim.x + threadIdx.x;
    int stride = gridDim.x * blockDim.x;
    for (; i < n; i += stride) out[i] = f32_to_bf16(in[i]);
}

// ---------------------------------------------------------------------------
// NT bf16 GEMM:  C[M,N] = A[M,K] @ W[N,K]^T + bias
//   MODE 0: write fp32 row-major to outF
//   MODE 1: write bf16 to outH in head-split [B,H,S,64] layout
// Block tile 128x128, 8 waves in 4x2, each wave 32x64 (2x4 WMMA accumulators).
// K-tile 32, DOUBLE-BUFFERED in LDS via async global->LDS copies.
// ---------------------------------------------------------------------------
template <int MODE>
__global__ __launch_bounds__(256)
void gemm_nt_bf16_kernel(const unsigned short* __restrict__ A,
                         const unsigned short* __restrict__ Wm,
                         const float* __restrict__ bias,
                         float* __restrict__ outF,
                         unsigned short* __restrict__ outH) {
    __shared__ unsigned short As[2][128][40];
    __shared__ unsigned short Bs[2][128][40];

    const int tid  = threadIdx.x;
    const int lane = tid & 31;
    const int wid  = tid >> 5;
    const int hl   = lane >> 4;   // lane half (0/1)
    const int ln   = lane & 15;   // lane within half
    const int m0   = blockIdx.y * 128;
    const int n0   = blockIdx.x * 128;
    const int wm   = wid >> 1;    // 0..3 : 32-row chunk
    const int wn   = wid & 1;     // 0..1 : 64-col chunk

    const v8f zero8 = {0.f,0.f,0.f,0.f,0.f,0.f,0.f,0.f};
    v8f acc[2][4];
#pragma unroll
    for (int i = 0; i < 2; ++i)
#pragma unroll
        for (int j = 0; j < 4; ++j) acc[i][j] = zero8;

    const int lrow  = tid >> 1;   // 0..127
    const int lhalf = tid & 1;    // 0..1
    const unsigned short* gA = A  + (size_t)(m0 + lrow) * GK + lhalf * 16;
    const unsigned short* gB = Wm + (size_t)(n0 + lrow) * GK + lhalf * 16;
    unsigned short* lA0 = &As[0][lrow][lhalf * 16];
    unsigned short* lB0 = &Bs[0][lrow][lhalf * 16];
    unsigned short* lA1 = &As[1][lrow][lhalf * 16];
    unsigned short* lB1 = &Bs[1][lrow][lhalf * 16];

    // prologue: stage tile 0 into buffer 0
    async_copy_32B(gA, lA0);
    async_copy_32B(gB, lB0);
    wait_asynccnt0();
    __syncthreads();

    for (int kt = 0; kt < GK / 32; ++kt) {
        const int cur = kt & 1;
        // issue async copies for the next tile into the other buffer
        if (kt + 1 < GK / 32) {
            async_copy_32B(gA + (kt + 1) * 32, cur ? lA0 : lA1);
            async_copy_32B(gB + (kt + 1) * 32, cur ? lB0 : lB1);
        }

        // A fragments (16x32, ISA 16-bit A layout: lane=M, per-lane K runs of 8)
        v16bf af[2];
#pragma unroll
        for (int ms = 0; ms < 2; ++ms) {
            const unsigned short* p = &As[cur][wm * 32 + ms * 16 + ln][0];
            V16U r;
            r.q[0] = *(const uint4*)(p + hl * 8);
            r.q[1] = *(const uint4*)(p + 16 + hl * 8);
            af[ms] = r.v;
        }
        // B fragments (32x16, same K order as A: lane=N, per-lane K runs of 8)
        v16bf bfr[4];
#pragma unroll
        for (int ns = 0; ns < 4; ++ns) {
            const unsigned short* p = &Bs[cur][wn * 64 + ns * 16 + ln][0];
            V16U r;
            r.q[0] = *(const uint4*)(p + hl * 8);
            r.q[1] = *(const uint4*)(p + 16 + hl * 8);
            bfr[ns] = r.v;
        }
#pragma unroll
        for (int ms = 0; ms < 2; ++ms)
#pragma unroll
            for (int ns = 0; ns < 4; ++ns)
                acc[ms][ns] = __builtin_amdgcn_wmma_f32_16x16x32_bf16(
                    false, af[ms], false, bfr[ns], (short)0, acc[ms][ns], false, false);

        wait_asynccnt0();   // next tile resident
        __syncthreads();    // all waves done with current tile
    }

    // store (C layout: VGPR r -> row r + 8*hl, lane ln -> col)
#pragma unroll
    for (int ms = 0; ms < 2; ++ms) {
        const int mrow = m0 + wm * 32 + ms * 16 + 8 * hl;
#pragma unroll
        for (int ns = 0; ns < 4; ++ns) {
            const int col = n0 + wn * 64 + ns * 16 + ln;
            const float bv = bias[col];
#pragma unroll
            for (int r = 0; r < 8; ++r) {
                const int m = mrow + r;
                const float v = acc[ms][ns][r] + bv;
                if constexpr (MODE == 0) {
                    outF[(size_t)m * GN + col] = v;
                } else {
                    const int b = m >> 11;     // m / S
                    const int s = m & 2047;    // m % S
                    const int h = col >> 6;    // col / 64
                    const int d = col & 63;
                    outH[(((size_t)(b * HH + h)) * SS + s) * ND + d] = f32_to_bf16(v);
                }
            }
        }
    }
}

// ---------------------------------------------------------------------------
// Causal flash attention, bf16 WMMA, fp32 accumulate + online softmax.
// Grid: (16 q-tiles of 128 rows, 32 batch*head). Block: 256 threads = 8 waves,
// each wave owns 16 query rows. K tile row-major (async global->LDS),
// V tile transposed in LDS so P@V B-fragments are contiguous reads.
// P converted C-layout -> A-layout through per-wave LDS (same-wave DS order).
// ---------------------------------------------------------------------------
__global__ __launch_bounds__(256)
void attn_kernel(const unsigned short* __restrict__ Qh,
                 const unsigned short* __restrict__ Kh,
                 const unsigned short* __restrict__ Vh,
                 unsigned short* __restrict__ Ao) {
    __shared__ unsigned short Ks[64][72];      // [key][d]
    __shared__ unsigned short Vt[64][72];      // [d][key] (transposed)
    __shared__ unsigned short Pw[8][16][72];   // per-wave P tile 16x64

    const int tid  = threadIdx.x;
    const int lane = tid & 31;
    const int wid  = tid >> 5;
    const int hl   = lane >> 4;
    const int ln   = lane & 15;
    const int qt   = blockIdx.x;        // 0..15
    const int bh   = blockIdx.y;        // 0..31
    const int q0   = qt * 128;
    const int qbase = q0 + wid * 16;
    const size_t base = (size_t)bh * SS * ND;

    // Q fragments (registers, reused for all key tiles)
    v16bf qf[2];
    {
        const unsigned short* qp = Qh + base + (size_t)(qbase + ln) * ND;
#pragma unroll
        for (int ks = 0; ks < 2; ++ks) {
            V16U r;
            r.q[0] = *(const uint4*)(qp + ks * 32 + hl * 8);
            r.q[1] = *(const uint4*)(qp + ks * 32 + 16 + hl * 8);
            qf[ks] = r.v;
        }
    }

    float m_cur[8], l_cur[8];
#pragma unroll
    for (int r = 0; r < 8; ++r) { m_cur[r] = -__builtin_inff(); l_cur[r] = 0.f; }
    const v8f zero8 = {0.f,0.f,0.f,0.f,0.f,0.f,0.f,0.f};
    v8f oacc[4];
#pragma unroll
    for (int d = 0; d < 4; ++d) oacc[d] = zero8;

    const float scale = 0.125f;          // 1/sqrt(64)
    const int nkt = q0 / 64 + 2;         // causal: keys up to q0+127

    const int key = tid >> 2;            // 0..63
    const int dc  = (tid & 3) * 16;      // 0,16,32,48
    const unsigned short* gkbase = Kh + base + (size_t)key * ND + dc;
    const unsigned short* gvbase = Vh + base + (size_t)key * ND + dc;

    for (int kt = 0; kt < nkt; ++kt) {
        const int kb = kt * 64;
        // K tile: async global->LDS.  V tile: register transpose into LDS.
        async_copy_32B(gkbase + (size_t)kb * ND, &Ks[key][dc]);
        {
            const unsigned short* gv = gvbase + (size_t)kb * ND;
#pragma unroll
            for (int e = 0; e < 16; ++e) Vt[dc + e][key] = gv[e];
        }
        if (kt + 1 < nkt)   // prefetch next V tile into cache
            __builtin_prefetch(gvbase + (size_t)(kb + 64) * ND, 0, 0);
        wait_asynccnt0();
        __syncthreads();

        // scores: S = Q (16x64) @ K^T  -> 4 fragments of 16x16
        v8f sf[4];
#pragma unroll
        for (int ns = 0; ns < 4; ++ns) {
            sf[ns] = zero8;
            const unsigned short* p = &Ks[ns * 16 + ln][0];
#pragma unroll
            for (int ks = 0; ks < 2; ++ks) {
                V16U r;
                r.q[0] = *(const uint4*)(p + ks * 32 + hl * 8);
                r.q[1] = *(const uint4*)(p + ks * 32 + 16 + hl * 8);
                sf[ns] = __builtin_amdgcn_wmma_f32_16x16x32_bf16(
                    false, qf[ks], false, r.v, (short)0, sf[ns], false, false);
            }
        }

        // scale + causal mask (reference: masked -> -1e9)
        float sc[4][8];
#pragma unroll
        for (int ns = 0; ns < 4; ++ns) {
            const int col = kb + ns * 16 + ln;
#pragma unroll
            for (int r = 0; r < 8; ++r) {
                const int row = qbase + 8 * hl + r;
                const float s = sf[ns][r] * scale;
                sc[ns][r] = (col > row) ? -1e9f : s;
            }
        }

        // online softmax (row stats live redundantly in each 16-lane group)
#pragma unroll
        for (int r = 0; r < 8; ++r) {
            float rm = fmaxf(fmaxf(sc[0][r], sc[1][r]), fmaxf(sc[2][r], sc[3][r]));
#pragma unroll
            for (int off = 1; off < 16; off <<= 1)
                rm = fmaxf(rm, __shfl_xor(rm, off, 32));
            const float mn = fmaxf(m_cur[r], rm);
            const float alpha = __expf(m_cur[r] - mn);
            float psum = 0.f;
#pragma unroll
            for (int ns = 0; ns < 4; ++ns) {
                const float p = __expf(sc[ns][r] - mn);
                sc[ns][r] = p;
                psum += p;
            }
#pragma unroll
            for (int off = 1; off < 16; off <<= 1)
                psum += __shfl_xor(psum, off, 32);
            l_cur[r] = l_cur[r] * alpha + psum;
            m_cur[r] = mn;
#pragma unroll
            for (int d = 0; d < 4; ++d) oacc[d][r] *= alpha;
        }

        // P -> bf16 in per-wave LDS (C layout write)
#pragma unroll
        for (int ns = 0; ns < 4; ++ns)
#pragma unroll
            for (int r = 0; r < 8; ++r)
                Pw[wid][8 * hl + r][ns * 16 + ln] = f32_to_bf16(sc[ns][r]);

        // re-read P as A fragments (same-wave LDS ordering)
        V16U pa[2];
        {
            const unsigned short* pp = &Pw[wid][ln][0];
#pragma unroll
            for (int ks = 0; ks < 2; ++ks) {
                pa[ks].q[0] = *(const uint4*)(pp + ks * 32 + hl * 8);
                pa[ks].q[1] = *(const uint4*)(pp + ks * 32 + 16 + hl * 8);
            }
        }

        // O += P (16x64) @ V (64x64)
#pragma unroll
        for (int ds = 0; ds < 4; ++ds) {
            const int d = ds * 16 + ln;
            const unsigned short* vp = &Vt[d][0];
#pragma unroll
            for (int ks = 0; ks < 2; ++ks) {
                V16U b;
                b.q[0] = *(const uint4*)(vp + ks * 32 + hl * 8);
                b.q[1] = *(const uint4*)(vp + ks * 32 + 16 + hl * 8);
                oacc[ds] = __builtin_amdgcn_wmma_f32_16x16x32_bf16(
                    false, pa[ks].v, false, b.v, (short)0, oacc[ds], false, false);
            }
        }
        __syncthreads();
    }

    // normalize and write attention output in merged [B,S,D] bf16 layout
    const int b = bh >> 4, h = bh & 15;
#pragma unroll
    for (int ds = 0; ds < 4; ++ds)
#pragma unroll
        for (int r = 0; r < 8; ++r) {
            const int qrow = qbase + 8 * hl + r;
            const float v = oacc[ds][r] / l_cur[r];
            Ao[((size_t)(b * SS + qrow)) * DD + h * ND + ds * 16 + ln] = f32_to_bf16(v);
        }
}

// ---------------------------------------------------------------------------
// Launch
// ---------------------------------------------------------------------------
extern "C" void kernel_launch(void* const* d_in, const int* in_sizes, int n_in,
                              void* d_out, int out_size, void* d_ws, size_t ws_size,
                              hipStream_t stream) {
    (void)in_sizes; (void)n_in; (void)out_size; (void)ws_size;
    const float* q  = (const float*)d_in[0];
    const float* k  = (const float*)d_in[1];
    const float* v  = (const float*)d_in[2];
    // d_in[3] = causal mask (handled analytically)
    const float* Wq = (const float*)d_in[4];
    const float* bq = (const float*)d_in[5];
    const float* Wk = (const float*)d_in[6];
    const float* bk = (const float*)d_in[7];
    const float* Wv = (const float*)d_in[8];
    const float* bv = (const float*)d_in[9];
    const float* Wo = (const float*)d_in[10];
    const float* bo = (const float*)d_in[11];

    unsigned short* ws = (unsigned short*)d_ws;
    const size_t E = (size_t)GM * GN;     // 4M elems
    const size_t W = (size_t)GN * GK;     // 1M elems
    unsigned short* qB  = ws;
    unsigned short* kB  = ws + E;
    unsigned short* vB  = ws + 2 * E;
    unsigned short* wqB = ws + 3 * E;
    unsigned short* wkB = wqB + W;
    unsigned short* wvB = wkB + W;
    unsigned short* woB = wvB + W;
    unsigned short* Qh  = woB + W;
    unsigned short* Kh  = Qh + E;
    unsigned short* Vh  = Kh + E;
    unsigned short* Ao  = Vh + E;

    cvt_bf16_kernel<<<1024, 256, 0, stream>>>(q,  qB,  (int)E);
    cvt_bf16_kernel<<<1024, 256, 0, stream>>>(k,  kB,  (int)E);
    cvt_bf16_kernel<<<1024, 256, 0, stream>>>(v,  vB,  (int)E);
    cvt_bf16_kernel<<<256,  256, 0, stream>>>(Wq, wqB, (int)W);
    cvt_bf16_kernel<<<256,  256, 0, stream>>>(Wk, wkB, (int)W);
    cvt_bf16_kernel<<<256,  256, 0, stream>>>(Wv, wvB, (int)W);
    cvt_bf16_kernel<<<256,  256, 0, stream>>>(Wo, woB, (int)W);

    dim3 gg(GN / 128, GM / 128);   // 8 x 32
    gemm_nt_bf16_kernel<1><<<gg, 256, 0, stream>>>(qB, wqB, bq, nullptr, Qh);
    gemm_nt_bf16_kernel<1><<<gg, 256, 0, stream>>>(kB, wkB, bk, nullptr, Kh);
    gemm_nt_bf16_kernel<1><<<gg, 256, 0, stream>>>(vB, wvB, bv, nullptr, Vh);

    attn_kernel<<<dim3(16, 32), 256, 0, stream>>>(Qh, Kh, Vh, Ao);

    gemm_nt_bf16_kernel<0><<<gg, 256, 0, stream>>>(Ao, woB, bo, (float*)d_out, nullptr);
}
